// My_Attn_17506286698697
// MI455X (gfx1250) — compile-verified, matrix operand order checked by reading, not run on hardware
//
#include <hip/hip_runtime.h>
#include <hip/hip_bf16.h>

typedef __bf16 bf16_t;
typedef __attribute__((ext_vector_type(16))) __bf16 v16bf;
typedef __attribute__((ext_vector_type(8)))  __bf16 v8bf;
typedef __attribute__((ext_vector_type(8)))  float  v8f;
typedef __attribute__((ext_vector_type(4)))  int    v4i;

#if __has_builtin(__builtin_amdgcn_global_load_async_to_lds_b128)
#define HAVE_ASYNC_LDS 1
#else
#define HAVE_ASYNC_LDS 0
#endif

static __device__ __forceinline__ bf16_t f2bf(float f) {
  unsigned u = __builtin_bit_cast(unsigned, f);
  unsigned r = (u + 0x7FFFu + ((u >> 16) & 1u)) >> 16;   // round-to-nearest-even
  unsigned short s = (unsigned short)r;
  return __builtin_bit_cast(bf16_t, s);
}

// 16 contiguous bf16 as a WMMA fragment; two 16B loads keep 16B alignment legal.
static __device__ __forceinline__ v16bf load16(const bf16_t* p) {
  union { v16bf v; v8bf h[2]; } u;
  u.h[0] = *(const v8bf*)(p);
  u.h[1] = *(const v8bf*)(p + 8);
  return u.v;
}
static __device__ __forceinline__ void store8(bf16_t* p, v8bf v) { *(v8bf*)p = v; }

#if HAVE_ASYNC_LDS
// 16B per lane, global -> LDS, tracked by ASYNCcnt (global_load_async_to_lds_b128)
static __device__ __forceinline__ void async_copy16(const bf16_t* g, bf16_t* l) {
  __builtin_amdgcn_global_load_async_to_lds_b128(
      (__attribute__((address_space(1))) v4i*)g,
      (__attribute__((address_space(3))) v4i*)l, 0, 0);
}
#endif

static __device__ __forceinline__ void wait_async0() {
#if __has_builtin(__builtin_amdgcn_s_wait_asynccnt)
  __builtin_amdgcn_s_wait_asynccnt(0);
#elif HAVE_ASYNC_LDS
  asm volatile("s_wait_asynccnt 0x0" ::: "memory");
#endif
}

static __device__ __forceinline__ v8f wmma_bf16(v16bf a, v16bf b, v8f c) {
  return __builtin_amdgcn_wmma_f32_16x16x32_bf16(false, a, false, b, (short)0, c,
                                                 false, false);
}

// ---------------------------------------------------------------------------
// f32 -> bf16 elementwise
__global__ __launch_bounds__(256)
void cvt_bf16_kern(const float* __restrict__ in, bf16_t* __restrict__ out, int n) {
  int i = blockIdx.x * 256 + threadIdx.x;
  if (i < n) out[i] = f2bf(in[i]);
}

// W[K,N] f32 -> Wt[N,K] bf16 (so GEMM B-fragments are row-contiguous)
__global__ void cvt_transpose_kern(const float* __restrict__ W,
                                   bf16_t* __restrict__ Wt, int K, int N) {
  int n = blockIdx.x * 16 + threadIdx.x;
  int k = blockIdx.y * 16 + threadIdx.y;
  if (n < N && k < K) Wt[(size_t)n * K + k] = f2bf(W[(size_t)k * N + n]);
}

// ---------------------------------------------------------------------------
// C[M,N] = A[M,K](bf16) @ Bt[N,K](bf16)^T + bias, f32 accumulate.
// 256 threads = 8 waves, macro tile 128(M) x 64(N), BK=32, waves 4x2 (32x32 each).
// Double-buffered LDS staging via async global->LDS copies overlapped with WMMA.
template <bool GELU, bool OUT_BF16>
__global__ __launch_bounds__(256)
void gemm_bf16_kern(const bf16_t* __restrict__ A, const bf16_t* __restrict__ Bt,
                    const float* __restrict__ bias, float* __restrict__ Cf,
                    bf16_t* __restrict__ Cb, int M, int N, int K) {
  __shared__ bf16_t As[2][128 * 40];  // 40-elem stride: 80B rows, 16B aligned
  __shared__ bf16_t Bs[2][64 * 40];
  const int tid  = threadIdx.x;
  const int lane = tid & 31;
  const int wave = tid >> 5;
  const int wm   = wave >> 1;  // 0..3
  const int wn   = wave & 1;   // 0..1
  const int m0   = blockIdx.y * 128;
  const int n0   = blockIdx.x * 64;
  const int mi   = lane & 15;
  const int kh   = lane >> 4;

  v8f acc[2][2] = {};

  const int ar = tid >> 1, ah = tid & 1;  // A tile: 128 rows x 2 halves of 16
  const int br = tid >> 2, bq = tid & 3;  // B tile: 64 rows x 4 quarters of 8

#if HAVE_ASYNC_LDS
  auto issue = [&](int k0, int buf) {
    const bf16_t* src = A + (size_t)(m0 + ar) * K + k0 + ah * 16;
    async_copy16(src,     &As[buf][ar * 40 + ah * 16]);
    async_copy16(src + 8, &As[buf][ar * 40 + ah * 16 + 8]);
    const bf16_t* srcb = Bt + (size_t)(n0 + br) * K + k0 + bq * 8;
    async_copy16(srcb, &Bs[buf][br * 40 + bq * 8]);
  };
  issue(0, 0);
#endif

  for (int k0 = 0, it = 0; k0 < K; k0 += 32, ++it) {
    const int cur = it & 1;
#if HAVE_ASYNC_LDS
    wait_async0();          // current tile resident in LDS
    __syncthreads();        // everyone done reading the other buffer
    if (k0 + 32 < K) issue(k0 + 32, cur ^ 1);   // overlap next tile with WMMA
#else
    __syncthreads();
    {
      const bf16_t* src = A + (size_t)(m0 + ar) * K + k0 + ah * 16;
      store8(&As[cur][ar * 40 + ah * 16],     *(const v8bf*)src);
      store8(&As[cur][ar * 40 + ah * 16 + 8], *(const v8bf*)(src + 8));
      const bf16_t* srcb = Bt + (size_t)(n0 + br) * K + k0 + bq * 8;
      store8(&Bs[cur][br * 40 + bq * 8], *(const v8bf*)srcb);
      if (k0 + 32 < K) __builtin_prefetch(src + 32, 0, 0);  // global_prefetch_b8
    }
    __syncthreads();
#endif
    v16bf af[2], bf[2];
#pragma unroll
    for (int i = 0; i < 2; ++i)
      af[i] = load16(&As[cur][(wm * 32 + i * 16 + mi) * 40 + kh * 16]);
#pragma unroll
    for (int j = 0; j < 2; ++j)
      bf[j] = load16(&Bs[cur][(wn * 32 + j * 16 + mi) * 40 + kh * 16]);
#pragma unroll
    for (int i = 0; i < 2; ++i)
#pragma unroll
      for (int j = 0; j < 2; ++j) acc[i][j] = wmma_bf16(af[i], bf[j], acc[i][j]);
  }

  // C/D layout: lane = (col, rowhalf); VGPR r = row r / r+8.
  const int rowb = kh * 8;
#pragma unroll
  for (int i = 0; i < 2; ++i) {
#pragma unroll
    for (int j = 0; j < 2; ++j) {
      const int n  = n0 + wn * 32 + j * 16 + mi;
      const float bv = bias ? bias[n] : 0.f;
#pragma unroll
      for (int r = 0; r < 8; ++r) {
        const int m = m0 + wm * 32 + i * 16 + rowb + r;
        float v = acc[i][j][r] + bv;
        if (GELU) v = 0.5f * v * (1.f + erff(v * 0.70710678118f));
        if (OUT_BF16) Cb[(size_t)m * N + n] = f2bf(v);
        else          Cf[(size_t)m * N + n] = v;
      }
    }
  }
}

// ---------------------------------------------------------------------------
// Flash attention over qkv bf16 [B*N, 3C] (q|k|v per row), H=16, D=64, N=1024.
// Grid (N/64, B*H); 128 threads = 4 waves; wave handles 16 query rows.
// K staged via async global->LDS; V staged transposed (manual scatter).
__global__ __launch_bounds__(128)
void attn_kern(const bf16_t* __restrict__ qkv, bf16_t* __restrict__ out) {
  __shared__ bf16_t Ks[32 * 72];       // [key][dim], 144B rows
  __shared__ bf16_t VsT[64 * 40];      // [dim][key], 80B rows
  __shared__ bf16_t Ps[4 * 16 * 40];   // per-wave P tile (C->A layout bounce)
  const int tid  = threadIdx.x;
  const int lane = tid & 31;
  const int wave = tid >> 5;
  const int b    = blockIdx.y >> 4;
  const int h    = blockIdx.y & 15;
  const int q0   = blockIdx.x * 64 + wave * 16;
  const int mi   = lane & 15;
  const int kh   = lane >> 4;
  const size_t rs = 3072;
  const bf16_t* base = qkv + (size_t)(b * 1024) * rs + h * 64;

  v16bf aq[2];
#pragma unroll
  for (int c = 0; c < 2; ++c)
    aq[c] = load16(base + (size_t)(q0 + mi) * rs + c * 32 + kh * 16);

  float m_run[8], l_run[8];
  v8f o_acc[4] = {};
#pragma unroll
  for (int r = 0; r < 8; ++r) { m_run[r] = -1e30f; l_run[r] = 0.f; }

  const int ck = tid >> 2, cs = tid & 3;   // cooperative stage indices
  bf16_t* psw = &Ps[wave * 16 * 40];

  for (int j0 = 0; j0 < 1024; j0 += 32) {
    __syncthreads();   // previous iteration's readers are done
    {
      const bf16_t* ksrc = base + (size_t)(j0 + ck) * rs + 1024 + cs * 16;
#if HAVE_ASYNC_LDS
      async_copy16(ksrc,     &Ks[ck * 72 + cs * 16]);
      async_copy16(ksrc + 8, &Ks[ck * 72 + cs * 16 + 8]);
#else
      store8(&Ks[ck * 72 + cs * 16],     *(const v8bf*)ksrc);
      store8(&Ks[ck * 72 + cs * 16 + 8], *(const v8bf*)(ksrc + 8));
#endif
      v16bf vv = load16(base + (size_t)(j0 + ck) * rs + 2048 + cs * 16);
#pragma unroll
      for (int i = 0; i < 16; ++i) VsT[(cs * 16 + i) * 40 + ck] = vv[i];
      wait_async0();
    }
    __syncthreads();

    // S(16x32) = Q(16x64) K^T : Ks rows are exactly Bt-layout for WMMA B.
    v8f s0 = {}, s1 = {};
#pragma unroll
    for (int c = 0; c < 2; ++c) {
      v16bf b0 = load16(&Ks[(mi)      * 72 + c * 32 + kh * 16]);
      v16bf b1 = load16(&Ks[(16 + mi) * 72 + c * 32 + kh * 16]);
      s0 = wmma_bf16(aq[c], b0, s0);
      s1 = wmma_bf16(aq[c], b1, s1);
    }

    // Online softmax per row slot r (row = r + 8*kh); row group = 16 lanes.
#pragma unroll
    for (int r = 0; r < 8; ++r) {
      float x0 = s0[r] * 0.125f, x1 = s1[r] * 0.125f;   // scale = 1/sqrt(64)
      float mx = fmaxf(x0, x1);
#pragma unroll
      for (int msk = 1; msk < 16; msk <<= 1) mx = fmaxf(mx, __shfl_xor(mx, msk, 32));
      float mnew = fmaxf(m_run[r], mx);
      float corr = __expf(m_run[r] - mnew);
      float p0 = __expf(x0 - mnew), p1 = __expf(x1 - mnew);
      float ps = p0 + p1;
#pragma unroll
      for (int msk = 1; msk < 16; msk <<= 1) ps += __shfl_xor(ps, msk, 32);
      l_run[r] = l_run[r] * corr + ps;
      m_run[r] = mnew;
#pragma unroll
      for (int dt = 0; dt < 4; ++dt) o_acc[dt][r] *= corr;
      const int rowl = r + kh * 8;
      psw[rowl * 40 + mi]      = f2bf(p0);
      psw[rowl * 40 + 16 + mi] = f2bf(p1);
    }
    __builtin_amdgcn_wave_barrier();   // keep DS store->load ordering tight

    // O(16x64) += P(16x32) V(32x64); VsT rows give contiguous B-fragments.
    v16bf pa = load16(&psw[mi * 40 + kh * 16]);
#pragma unroll
    for (int dt = 0; dt < 4; ++dt) {
      v16bf bv = load16(&VsT[(dt * 16 + mi) * 40 + kh * 16]);
      o_acc[dt] = wmma_bf16(pa, bv, o_acc[dt]);
    }
  }

  bf16_t* obase = out + (size_t)(b * 1024) * 1024 + h * 64;
#pragma unroll
  for (int dt = 0; dt < 4; ++dt)
#pragma unroll
    for (int r = 0; r < 8; ++r) {
      const int row = q0 + kh * 8 + r;
      obase[(size_t)row * 1024 + dt * 16 + mi] = f2bf(o_acc[dt][r] / l_run[r]);
    }
}

// ---------------------------------------------------------------------------
// out = res + LayerNorm(ln_in)*g + b  over C=1024; one row per workgroup.
__global__ __launch_bounds__(256)
void ln_residual_kern(const float* __restrict__ ln_in, const float* __restrict__ res,
                      const float* __restrict__ g, const float* __restrict__ bta,
                      float* __restrict__ outf, bf16_t* __restrict__ outb) {
  const int row = blockIdx.x, tid = threadIdx.x;
  const float* xr = ln_in + (size_t)row * 1024;
  __shared__ float red[256];
  float v[4], s = 0.f;
#pragma unroll
  for (int i = 0; i < 4; ++i) { v[i] = xr[tid + i * 256]; s += v[i]; }
  red[tid] = s; __syncthreads();
  for (int o = 128; o > 0; o >>= 1) { if (tid < o) red[tid] += red[tid + o]; __syncthreads(); }
  const float mu = red[0] * (1.f / 1024.f);
  __syncthreads();
  float s2 = 0.f;
#pragma unroll
  for (int i = 0; i < 4; ++i) { float d = v[i] - mu; s2 += d * d; }
  red[tid] = s2; __syncthreads();
  for (int o = 128; o > 0; o >>= 1) { if (tid < o) red[tid] += red[tid + o]; __syncthreads(); }
  const float rstd = rsqrtf(red[0] * (1.f / 1024.f) + 1e-5f);
#pragma unroll
  for (int i = 0; i < 4; ++i) {
    const int c = tid + i * 256;
    const float o = res[(size_t)row * 1024 + c] + (v[i] - mu) * rstd * g[c] + bta[c];
    outf[(size_t)row * 1024 + c] = o;
    if (outb) outb[(size_t)row * 1024 + c] = f2bf(o);
  }
}

// ---------------------------------------------------------------------------
extern "C" void kernel_launch(void* const* d_in, const int* in_sizes, int n_in,
                              void* d_out, int out_size, void* d_ws, size_t ws_size,
                              hipStream_t stream) {
  constexpr int Bb = 4, Nn = 1024, C = 1024, M = Bb * Nn;  // M = 4096
  const float* x_in   = (const float*)d_in[0];
  const float* qkv_w  = (const float*)d_in[1];
  const float* qkv_b  = (const float*)d_in[2];
  const float* proj_w = (const float*)d_in[3];
  const float* proj_b = (const float*)d_in[4];
  const float* n1_g   = (const float*)d_in[5];
  const float* n1_b   = (const float*)d_in[6];
  const float* n2_g   = (const float*)d_in[7];
  const float* n2_b   = (const float*)d_in[8];
  const float* fc1_w  = (const float*)d_in[9];
  const float* fc1_b  = (const float*)d_in[10];
  const float* fc2_w  = (const float*)d_in[11];
  const float* fc2_b  = (const float*)d_in[12];
  float* out = (float*)d_out;

  char* ws = (char*)d_ws;
  size_t off = 0;
  auto carve = [&](size_t bytes) -> char* {
    char* p = ws + off;
    off += (bytes + 255) & ~(size_t)255;
    return p;
  };
  bf16_t* xb     = (bf16_t*)carve((size_t)M * C * 2);      // x bf16, reused as pre_bf
  bf16_t* wqkvT  = (bf16_t*)carve((size_t)3 * C * C * 2);  // [3C, C]
  bf16_t* wprojT = (bf16_t*)carve((size_t)C * C * 2);      // [C, C]
  bf16_t* wfc1T  = (bf16_t*)carve((size_t)4 * C * C * 2);  // [4C, C]
  bf16_t* wfc2T  = (bf16_t*)carve((size_t)4 * C * C * 2);  // [C, 4C]
  bf16_t* qkvb   = (bf16_t*)carve((size_t)M * 4 * C * 2);  // qkv (24MB) / h1 (32MB)
  bf16_t* attn_o = (bf16_t*)carve((size_t)M * C * 2);
  float*  projo  = (float*)carve((size_t)M * C * 4);       // reused as h2
  float*  pre    = (float*)carve((size_t)M * C * 4);
  bf16_t* pre_bf = xb;     // xb dead after QKV GEMM
  bf16_t* h1     = qkvb;   // qkv dead after attention
  float*  h2     = projo;  // proj out dead after LN1

  cvt_bf16_kern<<<(M * C) / 256, 256, 0, stream>>>(x_in, xb, M * C);
  cvt_transpose_kern<<<dim3(3 * C / 16, C / 16), dim3(16, 16), 0, stream>>>(qkv_w, wqkvT, C, 3 * C);
  cvt_transpose_kern<<<dim3(C / 16, C / 16), dim3(16, 16), 0, stream>>>(proj_w, wprojT, C, C);
  cvt_transpose_kern<<<dim3(4 * C / 16, C / 16), dim3(16, 16), 0, stream>>>(fc1_w, wfc1T, C, 4 * C);
  cvt_transpose_kern<<<dim3(C / 16, 4 * C / 16), dim3(16, 16), 0, stream>>>(fc2_w, wfc2T, 4 * C, C);

  // QKV: [4096,1024] x [1024,3072]
  gemm_bf16_kern<false, true><<<dim3(3 * C / 64, M / 128), 256, 0, stream>>>(
      xb, wqkvT, qkv_b, nullptr, qkvb, M, 3 * C, C);
  // Attention: (N/64, B*H) workgroups
  attn_kern<<<dim3(Nn / 64, Bb * 16), 128, 0, stream>>>(qkvb, attn_o);
  // Proj: [4096,1024] x [1024,1024] -> f32
  gemm_bf16_kern<false, false><<<dim3(C / 64, M / 128), 256, 0, stream>>>(
      attn_o, wprojT, proj_b, projo, nullptr, M, C, C);
  // pre = x_in + LN1(proj)
  ln_residual_kern<<<M, 256, 0, stream>>>(projo, x_in, n1_g, n1_b, pre, pre_bf);
  // FC1 + GELU: [4096,1024] x [1024,4096] -> bf16
  gemm_bf16_kern<true, true><<<dim3(4 * C / 64, M / 128), 256, 0, stream>>>(
      pre_bf, wfc1T, fc1_b, nullptr, h1, M, 4 * C, C);
  // FC2: [4096,4096] x [4096,1024] -> f32
  gemm_bf16_kern<false, false><<<dim3(C / 64, M / 128), 256, 0, stream>>>(
      h1, wfc2T, fc2_b, h2, nullptr, M, C, 4 * C);
  // out = pre + LN2(h2)
  ln_residual_kern<<<M, 256, 0, stream>>>(h2, pre, n2_g, n2_b, out, nullptr);

  (void)in_sizes; (void)n_in; (void)out_size; (void)ws_size;
}
